// CrossAttn_30416958390902
// MI455X (gfx1250) — matrix operand order repaired and checked by reading, not compiled, not run on hardware
//
#include <hip/hip_runtime.h>
#include <hip/hip_bf16.h>
#include <math.h>

// ---------------------------------------------------------------------------
// CDNA5 (gfx1250, wave32) fused attention layer:
//   Q/K/V projections (WMMA f16, async-to-LDS staging) -> RoPE ->
//   flash attention (WMMA f16, exp2 softmax, WMMA row-sum) -> output proj
// ---------------------------------------------------------------------------

typedef _Float16 h16;
typedef __attribute__((ext_vector_type(16))) _Float16 v16h;
typedef __attribute__((ext_vector_type(8)))  _Float16 v8h;
typedef __attribute__((ext_vector_type(8)))  float    v8f;

static __device__ __forceinline__ v8f vzero8f() {
  v8f z = {0.f, 0.f, 0.f, 0.f, 0.f, 0.f, 0.f, 0.f};
  return z;
}

union Frag16 { v16h f; v8h h[2]; };

static __device__ __forceinline__ v16h frag_from2(const h16* p0, const h16* p1) {
  Frag16 u;
  u.h[0] = *(const v8h*)p0;
  u.h[1] = *(const v8h*)p1;
  return u.f;
}

// A-matrix fragment (16x32 MxK f16), per ISA 7.12.2.
static __device__ __forceinline__ v16h load_a_frag(const h16* base, int ld, int lane) {
  const int m  = lane & 15;
  const int kq = (lane >> 4) * 8;
  const h16* r = base + m * ld + kq;
  return frag_from2(r, r + 16);
}

// B-matrix fragment (32x16 KxN f16) loaded from B^T (N x K row-major, ld).
static __device__ __forceinline__ v16h load_b_frag(const h16* base, int ld, int lane) {
  const int n  = lane & 15;
  const int kq = (lane >> 4) * 8;
  const h16* r = base + n * ld + kq;
  return frag_from2(r, r + 16);
}

static __device__ __forceinline__ v8f wmma_f16(v16h a, v16h b, v8f c) {
  return __builtin_amdgcn_wmma_f32_16x16x32_f16(
      false, a, false, b, (short)0, c, false, false);
}

// ---------------------------------------------------------------------------
// CDNA5 async global->LDS DMA (ASYNCcnt-tracked), GVS addressing.
// ---------------------------------------------------------------------------
static __device__ __forceinline__ void async_b128(unsigned lds_off,
                                                  unsigned long long gbase,
                                                  unsigned goff) {
  asm volatile("global_load_async_to_lds_b128 %0, %1, %2"
               :: "v"(lds_off), "v"(goff), "s"(gbase) : "memory");
}

static __device__ __forceinline__ void wait_async0() {
  asm volatile("s_wait_asynccnt 0" ::: "memory");
}

static __device__ __forceinline__ unsigned lds_addr(const void* p) {
  return (unsigned)(unsigned long long)p;   // LDS offset = low 32 bits
}

// ---------------------------------------------------------------------------
// f32 -> f16 conversion, 8 elements / thread (b128-in/b128-out)
// ---------------------------------------------------------------------------
__global__ void cvt_f32_to_f16(const float* __restrict__ in, h16* __restrict__ out) {
  const int i = (blockIdx.x * blockDim.x + threadIdx.x) * 8;
  const float4 x0 = *(const float4*)(in + i);
  const float4 x1 = *(const float4*)(in + i + 4);
  v8h o;
  o[0] = (h16)x0.x; o[1] = (h16)x0.y; o[2] = (h16)x0.z; o[3] = (h16)x0.w;
  o[4] = (h16)x1.x; o[5] = (h16)x1.y; o[6] = (h16)x1.z; o[7] = (h16)x1.w;
  *(v8h*)(out + i) = o;
}

// ---------------------------------------------------------------------------
// GEMM: out[M,1024] = A[M,1024](f16) @ W[1024,1024]^T(f16) + bias.
// WG tile 64x256, 8 waves (2x4), 32x64/wave (2x4 WMMA subtiles).
// Double-buffered async-to-LDS staging.
// ---------------------------------------------------------------------------
template <int OUT_F32>
__global__ __launch_bounds__(256)
void gemm_wmma(const h16* __restrict__ A, const h16* __restrict__ W,
               const float* __restrict__ bias, void* __restrict__ outv)
{
  constexpr int N = 1024, K = 1024;
  __shared__ __align__(32) h16 As[2][64][40];     // 64 rows x 32 K
  __shared__ __align__(32) h16 Bs[2][256][40];    // 256 W-rows x 32 K

  const int tid  = threadIdx.x;
  const int lane = tid & 31;
  const int wave = tid >> 5;
  const int wr = wave >> 2;              // 0..1
  const int wc = wave & 3;               // 0..3
  const int m0 = blockIdx.y * 64;
  const int n0 = blockIdx.x * 256;

  v8f acc[2][4];
#pragma unroll
  for (int i = 0; i < 2; i++)
#pragma unroll
    for (int j = 0; j < 4; j++) acc[i][j] = vzero8f();

  const int a_row = tid >> 2, a_col = (tid & 3) * 8;   // 64x32: 1 b128/thread
  const int b_row = tid;                               // 256x32: 4 b128/thread

  const unsigned long long Ab = (unsigned long long)A;
  const unsigned long long Wb = (unsigned long long)W;
  const unsigned aoff = (unsigned)(((m0 + a_row) * K + a_col) * 2);
  const unsigned boff = (unsigned)(((n0 + b_row) * K) * 2);

  auto stage = [&](int buf, int kk) {
    async_b128(lds_addr(&As[buf][a_row][a_col]), Ab, aoff + kk * 2);
    const unsigned bl = lds_addr(&Bs[buf][b_row][0]);
    const unsigned bg = boff + kk * 2;
#pragma unroll
    for (int u = 0; u < 4; u++) async_b128(bl + u * 16, Wb, bg + u * 16);
  };

  stage(0, 0);
  for (int kk = 0, it = 0; kk < K; kk += 32, ++it) {
    const int buf = it & 1;
    wait_async0();                 // my DMAs into As/Bs[buf] have landed
    __syncthreads();               // everyone's have; prev readers of buf^1 done
    if (kk + 32 < K) stage(buf ^ 1, kk + 32);   // overlap next DMA with WMMAs

    v16h af[2], bf[4];
#pragma unroll
    for (int i = 0; i < 2; i++) af[i] = load_a_frag(&As[buf][wr * 32 + i * 16][0], 40, lane);
#pragma unroll
    for (int j = 0; j < 4; j++) bf[j] = load_b_frag(&Bs[buf][wc * 64 + j * 16][0], 40, lane);
#pragma unroll
    for (int i = 0; i < 2; i++)
#pragma unroll
      for (int j = 0; j < 4; j++)
        acc[i][j] = wmma_f16(af[i], bf[j], acc[i][j]);
  }

  // C layout: element e = (row = e + 8*(lane>>4), col = lane&15)
  const int hi = lane >> 4, lc = lane & 15;
#pragma unroll
  for (int i = 0; i < 2; i++) {
#pragma unroll
    for (int j = 0; j < 4; j++) {
      const int r0  = m0 + wr * 32 + i * 16 + hi * 8;
      const int col = n0 + wc * 64 + j * 16 + lc;
      const float bv = bias[col];
      if (OUT_F32) {
        float* po = (float*)outv + (size_t)r0 * N + col;
#pragma unroll
        for (int e = 0; e < 8; e++) po[(size_t)e * N] = acc[i][j][e] + bv;
      } else {
        h16* po = (h16*)outv + (size_t)r0 * N + col;
#pragma unroll
        for (int e = 0; e < 8; e++) po[(size_t)e * N] = (h16)(acc[i][j][e] + bv);
      }
    }
  }
}

// ---------------------------------------------------------------------------
// RoPE in place on [B*L, H*HD] f16; one thread owns pair (d, d+32).
// ---------------------------------------------------------------------------
__global__ void rope_f16(h16* __restrict__ X, const float* __restrict__ freqs)
{
  constexpr int L = 2048, H = 16, HD = 64, half = 32;
  const int idx = blockIdx.x * blockDim.x + threadIdx.x;
  const int d   = idx & (half - 1);
  const int t   = idx >> 5;
  const int hh  = t & (H - 1);
  const int row = t >> 4;                       // 0..B*L-1
  const int l   = row & (L - 1);
  const float fr = freqs[l * HD + d];
  float s, c;
  __sincosf(fr, &s, &c);
  const size_t base = (size_t)row * (H * HD) + hh * HD + d;
  const float x1 = (float)X[base];
  const float x2 = (float)X[base + half];
  X[base]        = (h16)(x1 * c - x2 * s);
  X[base + half] = (h16)(x2 * c + x1 * s);
}

// ---------------------------------------------------------------------------
// Flash attention, HD=64. WG = (b,h, 128 q rows); 8 waves x 16 q rows.
// exp2-domain softmax (scale*log2e folded into Q). Row sums via P @ ones on
// the matrix pipe (5th accumulator tile) -> no sum shuffles.
// ---------------------------------------------------------------------------
__global__ __launch_bounds__(256)
void attn_wmma(const h16* __restrict__ Qt, const h16* __restrict__ Kt,
               const h16* __restrict__ Vt, h16* __restrict__ Ot)
{
  constexpr int L = 2048, D = 1024, H = 16;
  __shared__ __align__(32) h16 Ks[2][64][72];   // rows l, cols d  (= B^T for QK^T)
  __shared__ __align__(32) h16 Vs[2][64][72];   // transposed: rows d, cols l
  __shared__ __align__(32) h16 Ps[8][16][72];   // per-wave P relayout scratch

  const int tid  = threadIdx.x;
  const int lane = tid & 31;
  const int wave = tid >> 5;
  const int b  = blockIdx.y >> 4;
  const int hh = blockIdx.y & (H - 1);
  const int q0 = blockIdx.x * 128 + wave * 16;
  const int colh = hh * 64;
  const int hi = lane >> 4, lc = lane & 15;

  // Q A-fragments; fold (1/sqrt(64)) * log2(e) so S is in exp2 domain.
  v16h aQ[2];
  {
    const int kq = hi * 8;
    const h16* p = Qt + (size_t)(b * L + q0 + lc) * D + colh;
    const h16 qs = (h16)(0.125f * 1.44269504088896f);
    aQ[0] = frag_from2(p + kq,      p + kq + 16) * qs;
    aQ[1] = frag_from2(p + 32 + kq, p + 32 + kq + 16) * qs;
  }

  // All-ones B fragment (layout-invariant): rowsum(P) = P @ ones.
  v16h vone;
#pragma unroll
  for (int x = 0; x < 16; x++) vone[x] = (h16)1.f;

  v8f oc[4], osum = vzero8f();
#pragma unroll
  for (int dt = 0; dt < 4; dt++) oc[dt] = vzero8f();
  float mrow[8];
#pragma unroll
  for (int i = 0; i < 8; i++) mrow[i] = -1e30f;

  const unsigned long long Kb = (unsigned long long)Kt;
  const int kr = tid >> 2, kc = (tid & 3) * 16;          // K block: 2 b128/thread

  auto stage = [&](int buf, int kb) {
    const unsigned kl = lds_addr(&Ks[buf][kr][kc]);
    const unsigned kg = (unsigned)((((b * L + kb + kr) * D) + colh + kc) * 2);
    async_b128(kl,      Kb, kg);
    async_b128(kl + 16, Kb, kg + 16);
#pragma unroll
    for (int u = 0; u < 2; u++) {
      const int unit = tid + u * 256;
      const int l  = unit >> 3;
      const int d8 = (unit & 7) * 8;
      const v8h vv = *(const v8h*)(Vt + (size_t)(b * L + kb + l) * D + colh + d8);
#pragma unroll
      for (int j = 0; j < 8; j++) Vs[buf][d8 + j][l] = vv[j];
    }
  };

  stage(0, 0);
  for (int kb = 0, it = 0; kb < L; kb += 64, ++it) {
    const int buf = it & 1;
    wait_async0();
    __syncthreads();
    if (kb + 64 < L) stage(buf ^ 1, kb + 64);   // overlap next K/V with compute

    // S2 = Q K^T (log2 domain). Chunk-0 of all tiles, then chunk-1, so each
    // dependent WMMA pair is >=4 matrix ops apart (avoids RAW nop quads).
    v8f sc[4];
#pragma unroll
    for (int ct = 0; ct < 4; ct++)
      sc[ct] = wmma_f16(aQ[0], load_b_frag(&Ks[buf][ct * 16][0], 72, lane), vzero8f());
#pragma unroll
    for (int ct = 0; ct < 4; ct++)
      sc[ct] = wmma_f16(aQ[1], load_b_frag(&Ks[buf][ct * 16][32], 72, lane), sc[ct]);

    // Online softmax in exp2 domain; row r = i + 8*hi lives in one half-wave.
    float scalef[8];
#pragma unroll
    for (int i = 0; i < 8; i++) {
      float mx = -1e30f;
#pragma unroll
      for (int ct = 0; ct < 4; ct++) mx = fmaxf(mx, sc[ct][i]);
      for (int off = 1; off < 16; off <<= 1) mx = fmaxf(mx, __shfl_xor(mx, off, 16));
      const float mn = fmaxf(mrow[i], mx);
      scalef[i] = __builtin_amdgcn_exp2f(mrow[i] - mn);
      mrow[i] = mn;
#pragma unroll
      for (int ct = 0; ct < 4; ct++) {
        const float p = __builtin_amdgcn_exp2f(sc[ct][i] - mn);
        Ps[wave][i + 8 * hi][ct * 16 + lc] = (h16)p;
      }
#pragma unroll
      for (int dt = 0; dt < 4; dt++) oc[dt][i] *= scalef[i];
      osum[i] *= scalef[i];
    }

    // O += P @ V ; denominator += P @ ones. Chunk-0 sweep then chunk-1 sweep.
    v16h aP[2];
    {
      const int kq = hi * 8;
      const h16* p = &Ps[wave][lc][0];
      aP[0] = frag_from2(p + kq,      p + kq + 16);
      aP[1] = frag_from2(p + 32 + kq, p + 32 + kq + 16);
    }
#pragma unroll
    for (int dt = 0; dt < 4; dt++)
      oc[dt] = wmma_f16(aP[0], load_b_frag(&Vs[buf][dt * 16][0], 72, lane), oc[dt]);
    osum = wmma_f16(aP[0], vone, osum);
#pragma unroll
    for (int dt = 0; dt < 4; dt++)
      oc[dt] = wmma_f16(aP[1], load_b_frag(&Vs[buf][dt * 16][32], 72, lane), oc[dt]);
    osum = wmma_f16(aP[1], vone, osum);
  }

  // Normalize (every column of osum holds its row's denominator) and store.
#pragma unroll
  for (int e = 0; e < 8; e++) {
    const float inv = 1.f / osum[e];
    const int row = q0 + e + 8 * hi;
    h16* po = Ot + (size_t)(b * L + row) * D + colh + lc;
#pragma unroll
    for (int dt = 0; dt < 4; dt++) po[dt * 16] = (h16)(oc[dt][e] * inv);
  }
}

// ---------------------------------------------------------------------------
// Launcher
// ---------------------------------------------------------------------------
extern "C" void kernel_launch(void* const* d_in, const int* in_sizes, int n_in,
                              void* d_out, int out_size, void* d_ws, size_t ws_size,
                              hipStream_t stream)
{
  (void)in_sizes; (void)n_in; (void)out_size; (void)ws_size;
  const int B = 2, L = 2048, D = 1024, H = 16, HD = 64;
  const int M = B * L;

  const float* q  = (const float*)d_in[0];
  const float* k  = (const float*)d_in[1];
  const float* v  = (const float*)d_in[2];
  const float* fr = (const float*)d_in[3];
  const float* Wq = (const float*)d_in[4];
  const float* bq = (const float*)d_in[5];
  const float* Wk = (const float*)d_in[6];
  const float* bk = (const float*)d_in[7];
  const float* Wv = (const float*)d_in[8];
  const float* bv = (const float*)d_in[9];
  const float* Wo = (const float*)d_in[10];
  const float* bo = (const float*)d_in[11];

  char* ws = (char*)d_ws;
  const size_t MB = 1024 * 1024;
  h16* Xq   = (h16*)(ws + 0);
  h16* Xk   = (h16*)(ws + 8  * MB);
  h16* Xv   = (h16*)(ws + 16 * MB);
  h16* Wq16 = (h16*)(ws + 24 * MB);
  h16* Wk16 = (h16*)(ws + 26 * MB);
  h16* Wv16 = (h16*)(ws + 28 * MB);
  h16* Wo16 = (h16*)(ws + 30 * MB);
  h16* Qt   = (h16*)(ws + 32 * MB);
  h16* Kt   = (h16*)(ws + 40 * MB);
  h16* Vt   = (h16*)(ws + 48 * MB);
  h16* Ot   = Xq;                         // Xq dead after Q projection

  const int nMD = M * D, nDD = D * D;

  cvt_f32_to_f16<<<nMD / 2048, 256, 0, stream>>>(q,  Xq);
  cvt_f32_to_f16<<<nMD / 2048, 256, 0, stream>>>(k,  Xk);
  cvt_f32_to_f16<<<nMD / 2048, 256, 0, stream>>>(v,  Xv);
  cvt_f32_to_f16<<<nDD / 2048, 256, 0, stream>>>(Wq, Wq16);
  cvt_f32_to_f16<<<nDD / 2048, 256, 0, stream>>>(Wk, Wk16);
  cvt_f32_to_f16<<<nDD / 2048, 256, 0, stream>>>(Wv, Wv16);
  cvt_f32_to_f16<<<nDD / 2048, 256, 0, stream>>>(Wo, Wo16);

  dim3 ggemm(D / 256, M / 64);            // (4, 64)
  gemm_wmma<0><<<ggemm, 256, 0, stream>>>(Xq, Wq16, bq, Qt);
  gemm_wmma<0><<<ggemm, 256, 0, stream>>>(Xk, Wk16, bk, Kt);
  gemm_wmma<0><<<ggemm, 256, 0, stream>>>(Xv, Wv16, bv, Vt);

  const int nrope = M * H * (HD / 2);
  rope_f16<<<nrope / 256, 256, 0, stream>>>(Qt, fr);
  rope_f16<<<nrope / 256, 256, 0, stream>>>(Kt, fr);

  attn_wmma<<<dim3(L / 128, B * H), 256, 0, stream>>>(Qt, Kt, Vt, Ot);

  gemm_wmma<1><<<ggemm, 256, 0, stream>>>(Ot, Wo16, bo, d_out);
}
